// PositionalGrammar_12463995093741
// MI455X (gfx1250) — matrix operand order repaired and checked by reading, not compiled, not run on hardware
//
#include <hip/hip_runtime.h>

typedef __attribute__((ext_vector_type(16))) _Float16 v16h;
typedef __attribute__((ext_vector_type(8)))  _Float16 v8h;
typedef __attribute__((ext_vector_type(8)))  float    v8f;

#define RNUM 6
#define BB   8
#define NN   1024
#define DD   128

// ---------------------------------------------------------------------------
// Kernel 0: f32 -> f16 conversion of symbols and W; out = symbols (base term).
// grid = BB*NN*DD/256 = 4096 blocks x 256 threads
// ---------------------------------------------------------------------------
__global__ __launch_bounds__(256)
void pg_prep(const float* __restrict__ sym, const float* __restrict__ W,
             float* __restrict__ out,
             _Float16* __restrict__ s16, _Float16* __restrict__ w16) {
  int t = blockIdx.x * 256 + threadIdx.x;          // covers BB*NN*DD exactly
  float sv = sym[t];
  s16[t] = (_Float16)sv;
  out[t] = sv;                                      // out = symbols + (agg later)
  if (t < RNUM * DD * DD) w16[t] = (_Float16)W[t];
}

// ---------------------------------------------------------------------------
// Kernel 1: T[r,b,i,e] = sum_k s[b,i,k] * W[r,e,k] + bias[r,e], stored
// transposed as f16: Tt[r][b][e][i]  (i contiguous -> aggregation B-frags
// are contiguous 32B runs).
// One wave per 16(i) x 16(e) tile; wave index in block = e-tile.
// grid = RNUM*BB*(NN/16) = 3072 blocks x 256 threads (8 waves)
// ---------------------------------------------------------------------------
__global__ __launch_bounds__(256)
void pg_transform(const _Float16* __restrict__ s16,
                  const _Float16* __restrict__ w16,
                  const float* __restrict__ bias,
                  _Float16* __restrict__ tt) {
  int lane = threadIdx.x & 31;
  int dt   = threadIdx.x >> 5;          // e-tile 0..7
  int blk  = blockIdx.x;
  int it   = blk & 63;                  // i-tile
  int b    = (blk >> 6) & 7;
  int r    = blk >> 9;
  int row  = lane & 15;
  int kh   = lane >> 4;                 // K-half select (contiguous k order used here)

  const _Float16* Ap = s16 + ((size_t)b * NN + (size_t)it * 16 + row) * DD;
  const _Float16* Bp = w16 + ((size_t)r * DD + (size_t)dt * 16 + row) * DD;

  v8f acc = {};
  #pragma unroll
  for (int k = 0; k < DD; k += 32) {
    // A-frag (contiguous k order, same as B): lane holds 16 contiguous K starting at kh*16
    v16h av = *(const v16h*)(Ap + k + kh * 16);
    // B-frag (same k order as A): lane holds 16 contiguous K starting at kh*16
    v16h bv = *(const v16h*)(Bp + k + kh * 16);
    acc = __builtin_amdgcn_wmma_f32_16x16x32_f16(false, av, false, bv,
                                                 (short)0, acc, false, false);
  }

  float bb = bias[r * DD + dt * 16 + row];
  v8h o;
  #pragma unroll
  for (int v = 0; v < 8; ++v) o[v] = (_Float16)(acc[v] + bb);

  // D layout: element v <-> (i = it*16 + kh*8 + v, e = dt*16 + row)
  _Float16* op = tt + (((size_t)r * BB + b) * DD + (size_t)dt * 16 + row) * NN
                    + (size_t)it * 16 + kh * 8;
  *(v8h*)op = o;
}

// ---------------------------------------------------------------------------
// Kernel 2: agg[b,j,:] += sum_r Mask_r^T @ T_r over this block's i-slice.
// Block = 8 waves sharing one i-slice; wave w owns j-tile jg*8+w. Each 32-i
// chunk of Tt (48 KB: [6r][128d][32i] f16) is staged Global->LDS ONCE per
// block via async-to-LDS DMA (ASYNCcnt), double-buffered so the next chunk's
// copy overlaps this chunk's 48 WMMAs. 8x less L2 traffic than per-wave loads.
// grid = BB * 8(jgroup) * 8(islice) = 512 blocks x 256 threads
// ---------------------------------------------------------------------------
__global__ __launch_bounds__(256)
void pg_aggregate(const float* __restrict__ pos,
                  const _Float16* __restrict__ tt,
                  float* __restrict__ out) {
  __shared__ __align__(64) _Float16 stage[2][RNUM * DD * 32];  // 2 x 48 KB

  int tid  = threadIdx.x;
  int lane = tid & 31;
  int w    = tid >> 5;                  // wave -> j-tile within group
  int b    = blockIdx.x >> 6;
  int jg   = (blockIdx.x >> 3) & 7;     // j-tile group of 8
  int s    = blockIdx.x & 7;            // i-slice (128 sources)
  int jt   = jg * 8 + w;
  int col  = lane & 15;
  int kh   = lane >> 4;
  int j    = jt * 16 + col;
  int i0   = s * 128;

  const float* pb = pos + (size_t)b * NN * 2;
  float pjx = pb[2 * j];
  float pjy = pb[2 * j + 1];

  const char* ttb = (const char*)tt + (size_t)b * DD * NN * 2;

  // Stage one 32-i chunk: 768 rows x 64B, flat-copied as 3072 16B segments.
  // Each of 256 threads issues 12 async b128 copies (ASYNCcnt-tracked).
  auto issue = [&](int buf, int ic) {
    unsigned lbase = (unsigned)(size_t)&stage[buf][0];
    #pragma unroll
    for (int q = 0; q < 12; ++q) {
      int g   = tid + 256 * q;          // segment id
      int row = g >> 2;                 // rr*128 + d
      int sub = (g & 3) * 16;           // byte offset within 64B row span
      const char* gp = ttb
          + (size_t)(row >> 7) * ((size_t)BB * DD * NN * 2)   // rr term
          + (size_t)(row & 127) * (NN * 2)                    // d term
          + (size_t)ic * 2 + sub;
      unsigned laddr = lbase + (unsigned)g * 16;
      asm volatile("global_load_async_to_lds_b128 %0, %1, off"
                   :: "v"(laddr), "v"(gp) : "memory");
    }
  };

  v8f acc[8] = {};

  issue(0, i0);
  #pragma unroll
  for (int c = 0; c < 4; ++c) {
    int buf = c & 1;
    if (c + 1 < 4) {
      issue(buf ^ 1, i0 + (c + 1) * 32);            // prefetch next chunk
      asm volatile("s_wait_asynccnt 0xc" ::: "memory");  // chunk c landed
    } else {
      asm volatile("s_wait_asynccnt 0x0" ::: "memory");
    }
    __syncthreads();                                 // visible to all waves

    int ic = i0 + c * 32;

    // Build six 0/1 mask A-fragments: A[j, i] = (rel(i,j)==r && i!=j)
    v16h A[RNUM];
    #pragma unroll
    for (int kk = 0; kk < 16; ++kk) {
      // f16 A layout: element kk -> K=kh*16+kk (contiguous, matches B)
      int i  = ic + kh * 16 + kk;
      float dx = pjx - pb[2 * i];
      float dy = pjy - pb[2 * i + 1];
      int r;
      if      (dy >  0.5f) r = 0;
      else if (dy < -0.5f) r = 1;
      else if (dx < -0.5f) r = 2;
      else if (dx >  0.5f) r = 3;
      else if (__builtin_fabsf(dx) < 0.3f && __builtin_fabsf(dy) < 0.3f) r = 4;
      else r = 5;
      int rm = (i != j) ? r : -1;       // exclude diagonal
      #pragma unroll
      for (int rr = 0; rr < RNUM; ++rr)
        A[rr][kk] = (rm == rr) ? (_Float16)1.0f : (_Float16)0.0f;
    }

    #pragma unroll
    for (int rr = 0; rr < RNUM; ++rr) {
      #pragma unroll
      for (int d = 0; d < 8; ++d) {
        // B-frag from LDS: lane col = d*16+col, 16 contiguous i from kh*16
        const _Float16* lp =
            &stage[buf][(size_t)(rr * DD + d * 16 + col) * 32 + kh * 16];
        v16h bv = *(const v16h*)lp;
        acc[d] = __builtin_amdgcn_wmma_f32_16x16x32_f16(false, A[rr], false, bv,
                                                        (short)0, acc[d],
                                                        false, false);
      }
    }
    __syncthreads();   // all waves done reading buf before it is re-filled
  }

  // Merge partials: element v of acc[d] <-> (j = jt*16 + kh*8 + v, e = d*16+col)
  #pragma unroll
  for (int d = 0; d < 8; ++d) {
    float* op = out + ((size_t)b * NN + (size_t)jt * 16 + kh * 8) * DD
                    + (size_t)d * 16 + col;
    #pragma unroll
    for (int v = 0; v < 8; ++v)
      atomicAdd(op + (size_t)v * DD, acc[d][v]);
  }
}

// ---------------------------------------------------------------------------
extern "C" void kernel_launch(void* const* d_in, const int* in_sizes, int n_in,
                              void* d_out, int out_size, void* d_ws, size_t ws_size,
                              hipStream_t stream) {
  (void)in_sizes; (void)n_in; (void)out_size; (void)ws_size;
  const float* symbols   = (const float*)d_in[0];   // [B,N,D] f32
  const float* positions = (const float*)d_in[1];   // [B,N,2] f32
  const float* W         = (const float*)d_in[2];   // [6,D,D] f32
  const float* bias      = (const float*)d_in[3];   // [6,D]   f32
  float* out = (float*)d_out;

  char* ws = (char*)d_ws;
  _Float16* s16 = (_Float16*)ws;                                   // 2 MB
  _Float16* w16 = (_Float16*)(ws + (2u << 20));                    // 192 KB
  _Float16* tt  = (_Float16*)(ws + (2u << 20) + (256u << 10));     // 12 MB

  pg_prep     <<<(BB * NN * DD) / 256, 256, 0, stream>>>(symbols, W, out, s16, w16);
  pg_transform<<<RNUM * BB * (NN / 16), 256, 0, stream>>>(s16, w16, bias, tt);
  pg_aggregate<<<BB * (NN / 16), 256, 0, stream>>>(positions, tt, out);
}
